// SnakeConvModel_44513041056357
// MI455X (gfx1250) — matrix operand order repaired and checked by reading, not compiled, not run on hardware
//
#include <hip/hip_runtime.h>
#include <hip/hip_bf16.h>

typedef __attribute__((ext_vector_type(16))) _Float16 v16h;
typedef __attribute__((ext_vector_type(8)))  _Float16 v8h;
typedef __attribute__((ext_vector_type(8)))  float    v8f;

#define NP 112           // padded spatial (100 -> 7*16)
#define LAYER_H 34816    // f16 elems per layer in workspace

__device__ __forceinline__ v16h pack16(v8h a, v8h b) {
  v16h r;
#pragma unroll
  for (int i = 0; i < 8; i++) { r[i] = a[i]; r[i + 8] = b[i]; }
  return r;
}

// A-operand: 16x32 f16 tile of W[out][K] (row-major, ld halfs), rows m..,
// K-chunk base k0.  Contiguous K order: lane (lh=lane>>4) holds
// K = k0 + lh*16 + i (i<16).
__device__ __forceinline__ v16h ldA(const _Float16* __restrict__ W, int ld,
                                    int m, int k0, int lh) {
  const _Float16* p = W + m * ld + k0 + lh * 16;
  return pack16(*(const v8h*)p, *(const v8h*)(p + 8));
}

// B-operand: 16 contiguous halfs (this lane's K-chunk of one column).
__device__ __forceinline__ v16h ldB(const _Float16* p) {
  return pack16(*(const v8h*)p, *(const v8h*)(p + 8));
}

__device__ __forceinline__ v8f wmma32(v16h a, v16h b, v8f c) {
  return __builtin_amdgcn_wmma_f32_16x16x32_f16(false, a, false, b,
                                                (short)0, c, false, false);
}

__device__ __forceinline__ v8f zero8() {
  v8f z = {0.f, 0.f, 0.f, 0.f, 0.f, 0.f, 0.f, 0.f};
  return z;
}

static __device__ void softmax_row(float* p, int n) {
  float mx = -1e30f;
  for (int i = 0; i < n; i++) mx = fmaxf(mx, p[i]);
  float s = 0.f;
  for (int i = 0; i < n; i++) { float e = __expf(p[i] - mx); p[i] = e; s += e; }
  float inv = 1.f / s;
  for (int i = 0; i < n; i++) p[i] *= inv;
}

// ---------------- prep: fp32 weights -> fp16 workspace ----------------
// layout per layer l (halfs, base l*LAYER_H):
//   [0,1024)      Watt   32x32
//   [1024,3072)   Wr1a   64x32
//   [3072,5120)   Wr1b   32x64
//   [5120,30720)  Wr2c   25 taps x (32x32)   (reordered from OIHW)
//   [30720,32768) Wr2a   64x32
//   [32768,34816) Wr2b   32x64
// heads at 8*LAYER_H: Wq1 128x32, then Wv1 128x32
__global__ void prep_kernel(const float* __restrict__ Watt,
                            const float* __restrict__ Wr1a,
                            const float* __restrict__ Wr1b,
                            const float* __restrict__ Wr2c,
                            const float* __restrict__ Wr2a,
                            const float* __restrict__ Wr2b,
                            const float* __restrict__ Wq1,
                            const float* __restrict__ Wv1,
                            _Float16* __restrict__ ws) {
  int idx = blockIdx.x * 256 + threadIdx.x;
  if (idx < 8 * LAYER_H) {
    int l = idx / LAYER_H, r = idx % LAYER_H;
    float v;
    if (r < 1024)       v = Watt[l * 1024 + r];
    else if (r < 3072)  v = Wr1a[l * 2048 + (r - 1024)];
    else if (r < 5120)  v = Wr1b[l * 2048 + (r - 3072)];
    else if (r < 30720) { int rr = r - 5120; int t = rr >> 10; int oi = rr & 1023;
                          v = Wr2c[l * 25600 + oi * 25 + t]; }
    else if (r < 32768) v = Wr2a[l * 2048 + (r - 30720)];
    else                v = Wr2b[l * 2048 + (r - 32768)];
    ws[idx] = (_Float16)v;
  } else if (idx < 8 * LAYER_H + 8192) {
    int r = idx - 8 * LAYER_H;
    float v = (r < 4096) ? Wq1[r] : Wv1[r - 4096];
    ws[idx] = (_Float16)v;
  }
}

// ---------------- main: one block per batch element ----------------
__global__ __launch_bounds__(256) void snake_fwd(
    const int* __restrict__ snake, const int* __restrict__ food,
    const float* __restrict__ pos_tab, const float* __restrict__ seq_tab,
    const float* __restrict__ food_tab,
    const float* __restrict__ Wwei, const float* __restrict__ Wval,
    const float* __restrict__ br1a, const float* __restrict__ br2a,
    const float* __restrict__ bq1, const float* __restrict__ Wq2,
    const float* __restrict__ Wqw, const float* __restrict__ bv1,
    const float* __restrict__ Wv2, const float* __restrict__ Wvw,
    const _Float16* __restrict__ ws, float* __restrict__ out, int nbatch) {
  __shared__ float xf[32 * NP];                      // residual stream [c][n]
  __shared__ __align__(16) _Float16 xT[NP * 32];     // f16 mirror [n][c]
  __shared__ __align__(16) _Float16 bT[NP * 32];     // xv / conv-out [n][c]
  __shared__ __align__(16) _Float16 hT[NP * 64];     // hidden [n][64] (aliases att16)
  __shared__ __align__(16) _Float16 padT[196 * 32];  // padded 14x14 board [p][c]
  __shared__ float wei[8 * NP];
  __shared__ float vtmp[32];
  __shared__ float vS[32];
  __shared__ float qacc[4 * NP];
  __shared__ float vacc[NP];

  const int tid = threadIdx.x;
  const int wv = tid >> 5;
  const int lane = tid & 31;
  const int lq = lane & 15;   // D-tile column offset
  const int lh = lane >> 4;   // half-lane group
  const int bidx = blockIdx.x;

  // ---- init: zero conv pad, embedding sum ----
  for (int i = tid; i < 196 * 32; i += 256) padT[i] = (_Float16)0.f;
  for (int i = tid; i < NP * 32; i += 256) {
    int n = i >> 5, c = i & 31;
    float v = 0.f;
    if (n < 100)
      v = seq_tab[snake[bidx * 100 + n] * 32 + c] +
          food_tab[food[bidx * 100 + n] * 32 + c] + pos_tab[n * 32 + c];
    xT[i] = (_Float16)v;
    xf[c * NP + n] = v;
  }
  __syncthreads();

  for (int l = 0; l < 8; l++) {
    const _Float16* wAtt = ws + l * LAYER_H;
    const _Float16* wR1a = wAtt + 1024;
    const _Float16* wR1b = wAtt + 3072;
    const _Float16* wR2c = wAtt + 5120;
    const _Float16* wR2a = wAtt + 30720;
    const _Float16* wR2b = wAtt + 32768;

    // ---- att = Watt @ x  (32x100), f16 into hT alias ----
    _Float16* att16 = hT;
    for (int t = wv; t < 14; t += 8) {
      int m0 = (t / 7) * 16, n0 = (t % 7) * 16;
      v16h A = ldA(wAtt, 32, m0 + lq, 0, lh);
      v16h Bv = ldB(xT + (n0 + lq) * 32 + lh * 16);
      v8f d = wmma32(A, Bv, zero8());
      int n = n0 + lq;
#pragma unroll
      for (int r = 0; r < 8; r++)
        att16[(m0 + lh * 8 + r) * NP + n] = (_Float16)d[r];
    }
    __syncthreads();

    // ---- wei logits (8x100) ----
    for (int i = tid; i < 800; i += 256) {
      int h = i / 100, n = i % 100;
      const float* ww = Wwei + l * 256 + h * 32;
      float s = 0.f;
#pragma unroll
      for (int c = 0; c < 32; c++) s += ww[c] * xf[c * NP + n];
      wei[h * NP + n] = s;
    }
    __syncthreads();
    // softmax over width only (per head, per row)
    if (tid < 80) softmax_row(wei + (tid / 10) * NP + (tid % 10) * 10, 10);
    __syncthreads();
    // v[c] = sum_n wei[c>>2][n]*att[c][n]
    if (tid < 32) {
      int c = tid; float s = 0.f;
      const float* wrow = wei + (c >> 2) * NP;
      const _Float16* arow = att16 + c * NP;
      for (int n = 0; n < 100; n++) s += wrow[n] * (float)arow[n];
      vtmp[c] = s;
    }
    __syncthreads();
    if (tid < 32) {
      const float* wr = Wval + l * 1024 + tid * 32;
      float s = 0.f;
#pragma unroll
      for (int c = 0; c < 32; c++) s += wr[c] * vtmp[c];
      vS[tid] = s;
    }
    __syncthreads();
    // xv = x + v  (f16, transposed for B-operand)
    for (int i = tid; i < NP * 32; i += 256) {
      int n = i >> 5, c = i & 31;
      float v = (n < 100) ? xf[c * NP + n] + vS[c] : 0.f;
      bT[i] = (_Float16)v;
    }
    __syncthreads();

    // ---- h1 = leaky(Wr1a@xv + b1a)  (64x100) ----
    for (int t = wv; t < 28; t += 8) {
      int m0 = (t / 7) * 16, n0 = (t % 7) * 16;
      v16h A = ldA(wR1a, 32, m0 + lq, 0, lh);
      int n = n0 + lq;
      v16h Bv = ldB(bT + n * 32 + lh * 16);
      v8f d = wmma32(A, Bv, zero8());
#pragma unroll
      for (int r = 0; r < 8; r++) {
        int m = m0 + lh * 8 + r;
        float hv = d[r] + br1a[l * 64 + m];
        hv = hv > 0.f ? hv : 0.01f * hv;
        hT[n * 64 + m] = (_Float16)hv;
      }
    }
    __syncthreads();

    // ---- r1 = Wr1b@h1 ; x2 = x + r1 -> padded board ----
    for (int t = wv; t < 14; t += 8) {
      int m0 = (t / 7) * 16, n0 = (t % 7) * 16;
      v16h A0 = ldA(wR1b, 64, m0 + lq, 0, lh);
      v16h A1 = ldA(wR1b, 64, m0 + lq, 32, lh);
      int n = n0 + lq;
      v16h B0 = ldB(hT + n * 64 + lh * 16);
      v16h B1 = ldB(hT + n * 64 + 32 + lh * 16);
      v8f d = wmma32(A0, B0, zero8());
      d = wmma32(A1, B1, d);
      if (n < 100) {
        int p = (n / 10 + 2) * 14 + (n % 10) + 2;
#pragma unroll
        for (int r = 0; r < 8; r++) {
          int m = m0 + lh * 8 + r;
          padT[p * 32 + m] = (_Float16)(xf[m * NP + n] + d[r]);
        }
      }
    }
    __syncthreads();

    // ---- c = conv5x5(x2): 25 shifted K=32 WMMA accumulations ----
    for (int t = wv; t < 14; t += 8) {
      int m0 = (t / 7) * 16, n0 = (t % 7) * 16;
      int n = n0 + lq;
      int nn = (n < 100) ? n : 0;  // clamp pad columns (discarded anyway)
      int pb = (nn / 10 + 2) * 14 + (nn % 10) + 2;
      v8f d = zero8();
      for (int tap = 0; tap < 25; tap++) {
        int off = (tap / 5 - 2) * 14 + (tap % 5 - 2);
        v16h A = ldA(wR2c + tap * 1024, 32, m0 + lq, 0, lh);
        v16h Bv = ldB(padT + (pb + off) * 32 + lh * 16);
        d = wmma32(A, Bv, d);
      }
#pragma unroll
      for (int r = 0; r < 8; r++)
        bT[n * 32 + m0 + lh * 8 + r] = (_Float16)d[r];
    }
    __syncthreads();

    // ---- h2 = leaky(Wr2a@c + b2a) ----
    for (int t = wv; t < 28; t += 8) {
      int m0 = (t / 7) * 16, n0 = (t % 7) * 16;
      v16h A = ldA(wR2a, 32, m0 + lq, 0, lh);
      int n = n0 + lq;
      v16h Bv = ldB(bT + n * 32 + lh * 16);
      v8f d = wmma32(A, Bv, zero8());
#pragma unroll
      for (int r = 0; r < 8; r++) {
        int m = m0 + lh * 8 + r;
        float hv = d[r] + br2a[l * 64 + m];
        hv = hv > 0.f ? hv : 0.01f * hv;
        hT[n * 64 + m] = (_Float16)hv;
      }
    }
    __syncthreads();

    // ---- x = x + Wr2b@h2  (residual uses pre-r1 stream) ----
    for (int t = wv; t < 14; t += 8) {
      int m0 = (t / 7) * 16, n0 = (t % 7) * 16;
      v16h A0 = ldA(wR2b, 64, m0 + lq, 0, lh);
      v16h A1 = ldA(wR2b, 64, m0 + lq, 32, lh);
      int n = n0 + lq;
      v16h B0 = ldB(hT + n * 64 + lh * 16);
      v16h B1 = ldB(hT + n * 64 + 32 + lh * 16);
      v8f d = wmma32(A0, B0, zero8());
      d = wmma32(A1, B1, d);
      if (n < 100) {
#pragma unroll
        for (int r = 0; r < 8; r++) {
          int m = m0 + lh * 8 + r;
          float nx = xf[m * NP + n] + d[r];
          xf[m * NP + n] = nx;
          xT[n * 32 + m] = (_Float16)nx;
        }
      }
    }
    __syncthreads();
  }

  // ---- output heads ----
  if (tid < NP) {
    int n = tid;
    qacc[n] = qacc[NP + n] = qacc[2 * NP + n] = qacc[3 * NP + n] = 0.f;
    vacc[n] = 0.f;
    if (n < 100) {
      float a = 0.f, b2 = 0.f;
#pragma unroll
      for (int c = 0; c < 32; c++) {
        float xv = xf[c * NP + n];
        a += Wqw[c] * xv; b2 += Wvw[c] * xv;
      }
      wei[n] = a; wei[NP + n] = b2;  // qw / vw logits
    }
  }
  __syncthreads();
  const _Float16* wQ1 = ws + 8 * LAYER_H;
  const _Float16* wV1 = wQ1 + 4096;
  for (int t = wv; t < 112; t += 8) {
    int isV = (t >= 56);
    int tt = isV ? t - 56 : t;
    int m0 = (tt / 7) * 16, n0 = (tt % 7) * 16;
    const _Float16* W = isV ? wV1 : wQ1;
    v16h A = ldA(W, 32, m0 + lq, 0, lh);
    int n = n0 + lq;
    v16h Bv = ldB(xT + n * 32 + lh * 16);
    v8f d = wmma32(A, Bv, zero8());
    if (n < 100) {
      if (!isV) {
        float p0 = 0.f, p1 = 0.f, p2 = 0.f, p3 = 0.f;
#pragma unroll
        for (int r = 0; r < 8; r++) {
          int m = m0 + lh * 8 + r;
          float hv = d[r] + bq1[m];
          hv = hv > 0.f ? hv : 0.01f * hv;
          p0 += Wq2[m] * hv;       p1 += Wq2[128 + m] * hv;
          p2 += Wq2[256 + m] * hv; p3 += Wq2[384 + m] * hv;
        }
        atomicAdd(&qacc[n], p0);          atomicAdd(&qacc[NP + n], p1);
        atomicAdd(&qacc[2 * NP + n], p2); atomicAdd(&qacc[3 * NP + n], p3);
      } else {
        float p = 0.f;
#pragma unroll
        for (int r = 0; r < 8; r++) {
          int m = m0 + lh * 8 + r;
          float hv = d[r] + bv1[m];
          hv = hv > 0.f ? hv : 0.01f * hv;
          p += Wv2[m] * hv;
        }
        atomicAdd(&vacc[n], p);
      }
    }
  }
  __syncthreads();
  if (tid == 0) softmax_row(wei, 100);
  else if (tid == 1) softmax_row(wei + NP, 100);
  __syncthreads();
  if (tid < 4) {
    float s = 0.f;
    for (int n = 0; n < 100; n++) s += qacc[tid * NP + n] * wei[n];
    out[bidx * 4 + tid] = s;
  } else if (tid == 4) {
    float s = 0.f;
    for (int n = 0; n < 100; n++) s += vacc[n] * wei[NP + n];
    out[nbatch * 4 + bidx] = s;
  }
}

extern "C" void kernel_launch(void* const* d_in, const int* in_sizes, int n_in,
                              void* d_out, int out_size, void* d_ws, size_t ws_size,
                              hipStream_t stream) {
  (void)n_in; (void)out_size; (void)ws_size;
  const int*   snake    = (const int*)d_in[0];
  const int*   food     = (const int*)d_in[1];
  const float* pos_tab  = (const float*)d_in[2];
  const float* seq_tab  = (const float*)d_in[3];
  const float* food_tab = (const float*)d_in[4];
  const float* Wwei = (const float*)d_in[5];
  const float* Watt = (const float*)d_in[6];
  const float* Wval = (const float*)d_in[7];
  const float* Wr1a = (const float*)d_in[8];
  const float* br1a = (const float*)d_in[9];
  const float* Wr1b = (const float*)d_in[10];
  const float* Wr2c = (const float*)d_in[11];
  const float* Wr2a = (const float*)d_in[12];
  const float* br2a = (const float*)d_in[13];
  const float* Wr2b = (const float*)d_in[14];
  const float* Wq1  = (const float*)d_in[15];
  const float* bq1  = (const float*)d_in[16];
  const float* Wq2  = (const float*)d_in[17];
  const float* Wqw  = (const float*)d_in[18];
  const float* Wv1  = (const float*)d_in[19];
  const float* bv1  = (const float*)d_in[20];
  const float* Wv2  = (const float*)d_in[21];
  const float* Wvw  = (const float*)d_in[22];
  float* out = (float*)d_out;
  _Float16* ws16 = (_Float16*)d_ws;

  int nbatch = in_sizes[0] / 100;

  int prep_total = 8 * LAYER_H + 8192;
  prep_kernel<<<(prep_total + 255) / 256, 256, 0, stream>>>(
      Watt, Wr1a, Wr1b, Wr2c, Wr2a, Wr2b, Wq1, Wv1, ws16);

  snake_fwd<<<nbatch, 256, 0, stream>>>(
      snake, food, pos_tab, seq_tab, food_tab, Wwei, Wval, br1a, br2a,
      bq1, Wq2, Wqw, bv1, Wv2, Wvw, ws16, out, nbatch);
}
